// PureSelfAttentionLayer_53764400611444
// MI455X (gfx1250) — compile-verified, matrix operand order checked by reading, not run on hardware
//
#include <hip/hip_runtime.h>
#include <hip/hip_bf16.h>
#include <stdint.h>

// ---------------------------------------------------------------------------
// CDNA5 (gfx1250) wave32 WMMA self-attention pipeline, bf16 compute / f32 accum
// Double-buffered operand fetch to overlap L2 latency with v_wmma issue.
// ---------------------------------------------------------------------------

typedef __attribute__((ext_vector_type(16))) __bf16 v16bf;
typedef __attribute__((ext_vector_type(8)))  float  v8f;

union Frag16 {
  v16bf    bf;
  uint32_t u[8];
  uint4    q[2];
};

__device__ __forceinline__ uint16_t f2bf(float x) {
  uint32_t u = __float_as_uint(x);
  uint32_t r = ((u >> 16) & 1u) + 0x7FFFu;   // round-to-nearest-even
  return (uint16_t)((u + r) >> 16);
}
__device__ __forceinline__ uint32_t pack2(float a, float b) {
  return (uint32_t)f2bf(a) | ((uint32_t)f2bf(b) << 16);
}

// 16x32 bf16 A/B operand from a row-major matrix (ld in elements).
// lane&15 = row; lanes<16 hold K=0..7 (v0-3), K=16..23 (v4-7);
// lanes>=16 hold K=8..15 and K=24..31.  (B: load B^T rows, same pattern.)
__device__ __forceinline__ void load_frag(Frag16& f, const uint16_t* base, int ld) {
  const int lane = threadIdx.x & 31;
  const uint16_t* p = base + (lane & 15) * ld + (lane >> 4) * 8;
  f.q[0] = *(const uint4*)(p);
  f.q[1] = *(const uint4*)(p + 16);
}

__device__ __forceinline__ v8f wmma_bf16(const Frag16& a, const Frag16& b, v8f c) {
  return __builtin_amdgcn_wmma_f32_16x16x32_bf16(false, a.bf, false, b.bf,
                                                 (short)0, c, false, false);
}

// ---------------------------------------------------------------------------
// Kernel 1: fp32 -> bf16 cast, vectorized 4-wide
// ---------------------------------------------------------------------------
__global__ void cast_bf16_x4(const float4* __restrict__ src,
                             uint2* __restrict__ dst, int n4) {
  int i = blockIdx.x * blockDim.x + threadIdx.x;
  int stride = gridDim.x * blockDim.x;
  for (; i < n4; i += stride) {
    float4 v = src[i];
    uint2 o;
    o.x = pack2(v.x, v.y);
    o.y = pack2(v.z, v.w);
    dst[i] = o;
  }
}

// ---------------------------------------------------------------------------
// Kernel 2: y = x @ W.T + b, head-split store.
//   vmode==0: dst[h][b][s][c]   (Q and K)
//   vmode==1: dst[h][b][c][s]   (V transposed -> contiguous b128 store)
// Block: 256 thr = 8 waves, tile 128m x 128n; wave = 32m x 64n.
// k-loop double-buffered: 12 b128 loads overlap 8 WMMAs.
// ---------------------------------------------------------------------------
__device__ __forceinline__ void gemm_load(Frag16 (&a)[2], Frag16 (&b)[4],
    const uint16_t* __restrict__ xb, const uint16_t* __restrict__ Wb,
    int mrow, int n0w, int k) {
  load_frag(a[0], xb + (mrow     ) * 1024 + k, 1024);
  load_frag(a[1], xb + (mrow + 16) * 1024 + k, 1024);
#pragma unroll
  for (int j = 0; j < 4; ++j)
    load_frag(b[j], Wb + (n0w + 16 * j) * 1024 + k, 1024);
}

__device__ __forceinline__ void gemm_step(const Frag16 (&a)[2],
    const Frag16 (&b)[4], v8f (&acc)[2][4]) {
#pragma unroll
  for (int i = 0; i < 2; ++i)
#pragma unroll
    for (int j = 0; j < 4; ++j)
      acc[i][j] = wmma_bf16(a[i], b[j], acc[i][j]);
}

__global__ void __launch_bounds__(256) qkv_gemm(
    const uint16_t* __restrict__ xb, const uint16_t* __restrict__ Wb,
    const float* __restrict__ bias, uint16_t* __restrict__ dst, int vmode) {
  const int wave  = threadIdx.x >> 5;
  const int lane  = threadIdx.x & 31;
  const int hi    = lane >> 4, l15 = lane & 15;
  const int mrow  = blockIdx.x * 128 + (wave & 3) * 32;
  const int n0w   = blockIdx.y * 128 + (wave >> 2) * 64;

  const v8f z = {0.f,0.f,0.f,0.f,0.f,0.f,0.f,0.f};
  v8f acc[2][4] = {{z,z,z,z},{z,z,z,z}};

  Frag16 a0[2], b0[4], a1[2], b1[4];
  gemm_load(a0, b0, xb, Wb, mrow, n0w, 0);
  for (int k = 0; k < 1024; k += 64) {
    gemm_load(a1, b1, xb, Wb, mrow, n0w, k + 32);        // prefetch stage B
    gemm_step(a0, b0, acc);
    gemm_load(a0, b0, xb, Wb, mrow, n0w, (k + 64) & 1023); // prefetch next A
    gemm_step(a1, b1, acc);
  }

#pragma unroll
  for (int i = 0; i < 2; ++i) {
    const int rrow  = mrow + 16 * i;
    const int bb    = rrow >> 10;      // batch; tiles never straddle b
    const int sbase = rrow & 1023;
#pragma unroll
    for (int j = 0; j < 4; ++j) {
      const int n  = n0w + 16 * j + l15;     // D: lane = N
      const int hh = n >> 6, ch = n & 63;    // never straddles heads
      const float bn = bias[n];
      if (vmode == 0) {
        uint16_t* d = dst + ((hh * 4 + bb) * 1024) * 64 + ch;
#pragma unroll
        for (int r = 0; r < 8; ++r) {        // D: VGPR r -> M = r + 8*hi
          int s = sbase + r + 8 * hi;
          d[s * 64] = f2bf(acc[i][j][r] + bn);
        }
      } else {                               // V^T: 8 consecutive s -> b128
        uint4 pk;
        pk.x = pack2(acc[i][j][0] + bn, acc[i][j][1] + bn);
        pk.y = pack2(acc[i][j][2] + bn, acc[i][j][3] + bn);
        pk.z = pack2(acc[i][j][4] + bn, acc[i][j][5] + bn);
        pk.w = pack2(acc[i][j][6] + bn, acc[i][j][7] + bn);
        uint16_t* d = dst + ((hh * 4 + bb) * 64 + ch) * 1024 + sbase + 8 * hi;
        *(uint4*)d = pk;
      }
    }
  }
}

// ---------------------------------------------------------------------------
// Kernel 3: softmax row stats over t: m_f = max_t(q_f.k_t/8), rZ_f = 1/sum.
// D[M=t, N=f]: A = K rows t, B^T = Q rows f. lane = f, VGPR = t.
// Double-buffered K tiles; lane-half combine via __shfl_xor(16).
// ---------------------------------------------------------------------------
__device__ __forceinline__ void stats_proc(const Frag16& ak0, const Frag16& ak1,
    const Frag16& bq0, const Frag16& bq1, float& m, float& Z) {
  const v8f z = {0.f,0.f,0.f,0.f,0.f,0.f,0.f,0.f};
  v8f sc = z;
  sc = wmma_bf16(ak0, bq0, sc);
  sc = wmma_bf16(ak1, bq1, sc);
  float tm = m;
#pragma unroll
  for (int r = 0; r < 8; ++r) tm = fmaxf(tm, sc[r] * 0.125f);
  Z *= __expf(m - tm);
#pragma unroll
  for (int r = 0; r < 8; ++r) Z += __expf(sc[r] * 0.125f - tm);
  m = tm;
}

__global__ void __launch_bounds__(256) softmax_stats(
    const uint16_t* __restrict__ qb, const uint16_t* __restrict__ kb,
    float* __restrict__ mOut, float* __restrict__ rZOut) {
  const int wave = threadIdx.x >> 5;
  const int lane = threadIdx.x & 31;
  const int hb   = blockIdx.y;                       // h*4 + b
  const int f0   = blockIdx.x * 128 + wave * 16;
  const uint16_t* qbase = qb + (size_t)hb * 1024 * 64;
  const uint16_t* kbase = kb + (size_t)hb * 1024 * 64;

  Frag16 bq0, bq1;                                   // hoisted B operands
  load_frag(bq0, qbase + f0 * 64 + 0,  64);
  load_frag(bq1, qbase + f0 * 64 + 32, 64);

  Frag16 k0a, k0b, k1a, k1b;
  load_frag(k0a, kbase + 0,  64);
  load_frag(k0b, kbase + 32, 64);

  float m = -3.0e38f, Z = 0.f;
  for (int t0 = 0; t0 < 1024; t0 += 32) {
    load_frag(k1a, kbase + (t0 + 16) * 64 + 0,  64);
    load_frag(k1b, kbase + (t0 + 16) * 64 + 32, 64);
    stats_proc(k0a, k0b, bq0, bq1, m, Z);
    const int tn = (t0 + 32) & 1023;
    load_frag(k0a, kbase + tn * 64 + 0,  64);
    load_frag(k0b, kbase + tn * 64 + 32, 64);
    stats_proc(k1a, k1b, bq0, bq1, m, Z);
  }
  float m2 = __shfl_xor(m, 16, 32);                  // other t-half, same f
  float Z2 = __shfl_xor(Z, 16, 32);
  float mm = fmaxf(m, m2);
  float Zt = Z * __expf(m - mm) + Z2 * __expf(m2 - mm);
  if (lane < 16) {
    mOut [hb * 1024 + f0 + lane] = mm;
    rZOut[hb * 1024 + f0 + lane] = 1.f / Zt;
  }
}

// ---------------------------------------------------------------------------
// Kernel 4: av^T[c,t] = sum_f V^T[c,f] * P[f,t];  out[b][t][c*16+h] (fp32).
// Score D[M=f, N=t] lands exactly in the next WMMA's B-operand layout after
// bf16 packing (lane = t both halves; VGPR r -> f = r + 8*hi): no transpose.
// Q fragments double-buffered across f-blocks; V loads issued before the
// exp transform so v_exp_f32 work hides their latency.
// ---------------------------------------------------------------------------
__device__ __forceinline__ void attn_load_q(Frag16 (&aq)[4],
    const uint16_t* __restrict__ qbase, int fb) {
  load_frag(aq[0], qbase + (fb     ) * 64 + 0,  64);
  load_frag(aq[1], qbase + (fb     ) * 64 + 32, 64);
  load_frag(aq[2], qbase + (fb + 16) * 64 + 0,  64);
  load_frag(aq[3], qbase + (fb + 16) * 64 + 32, 64);
}

__device__ __forceinline__ void attn_step(const Frag16 (&aq)[4],
    const Frag16& bk0, const Frag16& bk1,
    const uint16_t* __restrict__ vbase,
    const float* __restrict__ mrow, const float* __restrict__ zrow,
    int fb, int hi, v8f (&acc)[4]) {
  const v8f z = {0.f,0.f,0.f,0.f,0.f,0.f,0.f,0.f};
  v8f s1 = z, s2 = z;
  s1 = wmma_bf16(aq[0], bk0, s1);
  s1 = wmma_bf16(aq[1], bk1, s1);
  s2 = wmma_bf16(aq[2], bk0, s2);
  s2 = wmma_bf16(aq[3], bk1, s2);

  Frag16 av[4];                          // issue V loads before exp transform
#pragma unroll
  for (int j = 0; j < 4; ++j)
    load_frag(av[j], vbase + (16 * j) * 1024 + fb, 1024);

  Frag16 bp;                             // P as B operand [K=f(32), N=t(16)]
#pragma unroll
  for (int r = 0; r < 4; ++r) {
    int fA = fb + 2 * r + 8 * hi;
    float p0 = __expf(s1[2*r  ] * 0.125f - mrow[fA  ]) * zrow[fA  ];
    float p1 = __expf(s1[2*r+1] * 0.125f - mrow[fA+1]) * zrow[fA+1];
    bp.u[r] = pack2(p0, p1);
    int fB = fA + 16;
    float p2 = __expf(s2[2*r  ] * 0.125f - mrow[fB  ]) * zrow[fB  ];
    float p3 = __expf(s2[2*r+1] * 0.125f - mrow[fB+1]) * zrow[fB+1];
    bp.u[4 + r] = pack2(p2, p3);
  }
#pragma unroll
  for (int j = 0; j < 4; ++j)
    acc[j] = wmma_bf16(av[j], bp, acc[j]);
}

__global__ void __launch_bounds__(256) attention_av(
    const uint16_t* __restrict__ qb, const uint16_t* __restrict__ kb,
    const uint16_t* __restrict__ vtb, const float* __restrict__ mS,
    const float* __restrict__ rZ, float* __restrict__ out) {
  const int wave = threadIdx.x >> 5;
  const int lane = threadIdx.x & 31;
  const int hi   = lane >> 4, l15 = lane & 15;
  const int hb   = blockIdx.y;
  const int h    = hb >> 2, bb = hb & 3;
  const int t0   = blockIdx.x * 128 + wave * 16;
  const uint16_t* qbase = qb  + (size_t)hb * 1024 * 64;
  const uint16_t* kbase = kb  + (size_t)hb * 1024 * 64;
  const uint16_t* vbase = vtb + (size_t)hb * 64 * 1024;
  const float* mrow = mS + hb * 1024;
  const float* zrow = rZ + hb * 1024;

  Frag16 bk0, bk1;                       // B^T = K rows t, fixed for this wave
  load_frag(bk0, kbase + t0 * 64 + 0,  64);
  load_frag(bk1, kbase + t0 * 64 + 32, 64);

  const v8f z = {0.f,0.f,0.f,0.f,0.f,0.f,0.f,0.f};
  v8f acc[4] = {z, z, z, z};             // av^T: 4 c-tiles x one t-tile

  Frag16 q0[4], q1[4];
  attn_load_q(q0, qbase, 0);
  for (int fb = 0; fb < 1024; fb += 64) {
    attn_load_q(q1, qbase, fb + 32);                       // prefetch stage B
    attn_step(q0, bk0, bk1, vbase, mrow, zrow, fb, hi, acc);
    attn_load_q(q0, qbase, (fb + 64) & 1023);              // prefetch next A
    attn_step(q1, bk0, bk1, vbase, mrow, zrow, fb + 32, hi, acc);
  }

  const int t = t0 + l15;                // D: lane = N = t
#pragma unroll
  for (int j = 0; j < 4; ++j)
#pragma unroll
    for (int r = 0; r < 8; ++r) {
      int c = 16 * j + r + 8 * hi;       // D: VGPR r -> M = c
      out[(((size_t)bb * 1024 + t) * 64 + c) * 16 + h] = acc[j][r];
    }
}

// ---------------------------------------------------------------------------
extern "C" void kernel_launch(void* const* d_in, const int* in_sizes, int n_in,
                              void* d_out, int out_size, void* d_ws, size_t ws_size,
                              hipStream_t stream) {
  const float* x  = (const float*)d_in[0];
  const float* Wq = (const float*)d_in[1];
  const float* bq = (const float*)d_in[2];
  const float* Wk = (const float*)d_in[3];
  const float* bk = (const float*)d_in[4];
  const float* Wv = (const float*)d_in[5];
  const float* bv = (const float*)d_in[6];
  float* out = (float*)d_out;
  char*  ws  = (char*)d_ws;

  // workspace layout (bytes)
  uint16_t* xb   = (uint16_t*)(ws);                            //  8 MiB
  uint16_t* Wqb  = (uint16_t*)(ws + ( 8ull << 20));            //  2 MiB
  uint16_t* Wkb  = (uint16_t*)(ws + (10ull << 20));            //  2 MiB
  uint16_t* Wvb  = (uint16_t*)(ws + (12ull << 20));            //  2 MiB
  uint16_t* qbuf = (uint16_t*)(ws + (14ull << 20));            //  8 MiB [h,b,s,c]
  uint16_t* kbuf = (uint16_t*)(ws + (22ull << 20));            //  8 MiB [h,b,s,c]
  uint16_t* vbuf = (uint16_t*)(ws + (30ull << 20));            //  8 MiB [h,b,c,s]
  float*    mbuf = (float*)   (ws + (38ull << 20));            // 256 KiB
  float*    zbuf = (float*)   (ws + (38ull << 20) + (256u << 10));

  const int NX4 = (4 * 1024 * 1024) / 4;   // x float4 count
  const int NW4 = (1024 * 1024) / 4;       // per-weight float4 count

  cast_bf16_x4<<<dim3(2048), dim3(256), 0, stream>>>((const float4*)x,  (uint2*)xb,  NX4);
  cast_bf16_x4<<<dim3(1024), dim3(256), 0, stream>>>((const float4*)Wq, (uint2*)Wqb, NW4);
  cast_bf16_x4<<<dim3(1024), dim3(256), 0, stream>>>((const float4*)Wk, (uint2*)Wkb, NW4);
  cast_bf16_x4<<<dim3(1024), dim3(256), 0, stream>>>((const float4*)Wv, (uint2*)Wvb, NW4);

  dim3 ggrid(32, 8);                // 128m x 128n block tiles
  qkv_gemm<<<ggrid, dim3(256), 0, stream>>>(xb, Wqb, bq, qbuf, 0);
  qkv_gemm<<<ggrid, dim3(256), 0, stream>>>(xb, Wkb, bk, kbuf, 0);
  qkv_gemm<<<ggrid, dim3(256), 0, stream>>>(xb, Wvb, bv, vbuf, 1);

  dim3 agrid(8, 64);                // 128 rows per block x (h*b)
  softmax_stats<<<agrid, dim3(256), 0, stream>>>(qbuf, kbuf, mbuf, zbuf);
  attention_av <<<agrid, dim3(256), 0, stream>>>(qbuf, kbuf, vbuf, mbuf, zbuf, out);
}
